// GAT_28509992910944
// MI455X (gfx1250) — matrix operand order repaired and checked by reading, not compiled, not run on hardware
//
#include <hip/hip_runtime.h>
#include <math.h>

// ---------------------------------------------------------------------------
// GATv2 x3 + sum-readout + 2-layer MLP head for gfx1250 (MI455X, wave32).
// Dense node transforms: v_wmma_f32_16x16x32_f16, K templated so all tile
// loads are unconditional global_load_b128. Edge phase: L2-resident
// gather/scatter with native f32 atomics.
// ---------------------------------------------------------------------------

#define SLOPE 0.2f

typedef __attribute__((ext_vector_type(16))) _Float16 v16h;
typedef __attribute__((ext_vector_type(8)))  float    v8f;

// -------------------- WMMA node transform: Y = H @ W^T + bias --------------
// H: [nrows x K] row-major, W: [48 x K] row-major, Y: [nrows x 48].
// One wave per 16x16 output tile. Requires nrows % 16 == 0.
template <int K>
__global__ void gat_xform_wmma(const float* __restrict__ H,
                               const float* __restrict__ W,
                               const float* __restrict__ bias,
                               float* __restrict__ Y,
                               int mtiles) {
  int wave = (int)((blockIdx.x * blockDim.x + threadIdx.x) >> 5);
  int lane = (int)(threadIdx.x & 31);
  if (wave >= mtiles * 3) return;              // wave-uniform: EXEC stays full
  int mt = wave / 3;
  int nt = wave - mt * 3;
  int hi   = lane >> 4;                        // half-wave selector (0/1)
  int rowA = (mt << 4) + (lane & 15);          // A-matrix row for this lane
  int col  = (nt << 4) + (lane & 15);          // B/D column for this lane
  const float* __restrict__ hrow = H + (size_t)rowA * (size_t)K;
  const float* __restrict__ wrow = W + (size_t)col * (size_t)K;

  v8f c = {};
  #pragma unroll
  for (int kb = 0; kb < K; kb += 32) {
    v16h a = {};
    v16h b = {};
    // ---- A 16x32 f16: halves 0..7 -> k = kb + 8*hi + j (contiguous run)
    //                   halves 8..15 -> k = kb + 16 + 8*hi + j
    // Each 8-float run is entirely in range or entirely zero (compile-time).
    {
      const int k0 = kb;                       // + (hi<<3), handled below
      if (k0 + 8 + 8 <= K || true) {}          // keep compiler happy
      // run 1
      if (kb + 16 <= K) {                      // covers both hi halves
        const float4 p0 = *(const float4*)(hrow + kb + (hi << 3));
        const float4 p1 = *(const float4*)(hrow + kb + (hi << 3) + 4);
        a[0] = (_Float16)p0.x; a[1] = (_Float16)p0.y;
        a[2] = (_Float16)p0.z; a[3] = (_Float16)p0.w;
        a[4] = (_Float16)p1.x; a[5] = (_Float16)p1.y;
        a[6] = (_Float16)p1.z; a[7] = (_Float16)p1.w;
      }
      // run 2
      if (kb + 32 <= K) {                      // covers both hi halves
        const float4 p0 = *(const float4*)(hrow + kb + 16 + (hi << 3));
        const float4 p1 = *(const float4*)(hrow + kb + 16 + (hi << 3) + 4);
        a[8]  = (_Float16)p0.x; a[9]  = (_Float16)p0.y;
        a[10] = (_Float16)p0.z; a[11] = (_Float16)p0.w;
        a[12] = (_Float16)p1.x; a[13] = (_Float16)p1.y;
        a[14] = (_Float16)p1.z; a[15] = (_Float16)p1.w;
      }
    }
    // ---- B 32x16 f16: B[k][n] = W[n][k]; halves 0..7 -> k = kb + 8*hi + j,
    // halves 8..15 -> k = kb + 16 + 8*hi + j (same order as A)
    {
      if (kb + 16 <= K) {
        const float* wp = wrow + kb + (hi << 3);
        const float4 q0 = *(const float4*)(wp + 0);
        const float4 q1 = *(const float4*)(wp + 4);
        b[0]  = (_Float16)q0.x; b[1]  = (_Float16)q0.y;
        b[2]  = (_Float16)q0.z; b[3]  = (_Float16)q0.w;
        b[4]  = (_Float16)q1.x; b[5]  = (_Float16)q1.y;
        b[6]  = (_Float16)q1.z; b[7]  = (_Float16)q1.w;
      }
      if (kb + 32 <= K) {
        const float* wp = wrow + kb + 16 + (hi << 3);
        const float4 q2 = *(const float4*)(wp + 0);
        const float4 q3 = *(const float4*)(wp + 4);
        b[8]  = (_Float16)q2.x; b[9]  = (_Float16)q2.y;
        b[10] = (_Float16)q2.z; b[11] = (_Float16)q2.w;
        b[12] = (_Float16)q3.x; b[13] = (_Float16)q3.y;
        b[14] = (_Float16)q3.z; b[15] = (_Float16)q3.w;
      }
    }
    c = __builtin_amdgcn_wmma_f32_16x16x32_f16(false, a, false, b,
                                               (short)0, c, false, false);
  }
  float bn = bias[col];
  #pragma unroll
  for (int v = 0; v < 8; ++v) {                // D layout: M = v + 8*hi
    int row = (mt << 4) + v + (hi << 3);
    Y[(size_t)row * 48 + col] = c[v] + bn;
  }
}

// NOTE on A layout vs ISA table: A halves 0..7 hold K = base..base+7 with
// base = kb + 8*hi, halves 8..15 hold K = base+16..base+23. This matches the
// 16-bit A 16x32 table: lanes 0-15 VGPR0-3 = K0..7, VGPR4-7 = K16..23;
// lanes 16-31 shifted by +8.

// -------------------- per-layer init: out=0, emax=-inf, den=0 --------------
__global__ void gat_layer_init(float* __restrict__ out,
                               float* __restrict__ emax,
                               float* __restrict__ den, int n) {
  int i = blockIdx.x * blockDim.x + threadIdx.x;
  if (i < n * 48) out[i] = 0.0f;
  if (i < n) { emax[i] = -INFINITY; den[i] = 0.0f; }
}

__device__ __forceinline__ void atomicMaxF(float* addr, float val) {
  int* ia = (int*)addr;
  int old = __float_as_int(*addr);
  while (__int_as_float(old) < val) {
    int assumed = old;
    old = atomicCAS(ia, assumed, __float_as_int(val));
    if (old == assumed) break;
  }
}

// -------------------- edge pass A: logits + segment max --------------------
__global__ void gat_edge_logits(const int* __restrict__ src,
                                const int* __restrict__ dst,
                                const float* __restrict__ ea,
                                const float* __restrict__ xl,
                                const float* __restrict__ xr,
                                const float* __restrict__ We,   // [48 x 4]
                                const float* __restrict__ att,  // [48]
                                float* __restrict__ elog,
                                float* __restrict__ emax, int E) {
  __shared__ float sWe[48 * 4];
  __shared__ float sAtt[48];
  for (int i = threadIdx.x; i < 192; i += blockDim.x) sWe[i] = We[i];
  for (int i = threadIdx.x; i < 48;  i += blockDim.x) sAtt[i] = att[i];
  __syncthreads();
  int e = blockIdx.x * blockDim.x + threadIdx.x;
  if (e >= E) return;
  int s = src[e], d = dst[e];
  const float4 av = *(const float4*)(ea + (size_t)e * 4);
  const float4* pl = (const float4*)(xl + (size_t)s * 48);
  const float4* pr = (const float4*)(xr + (size_t)d * 48);
  __builtin_prefetch(pl, 0, 0);                // global_prefetch_b8
  __builtin_prefetch(pr, 0, 0);
  float logit = 0.0f;
  #pragma unroll
  for (int q = 0; q < 12; ++q) {               // 12 x float4 = 48 feats
    float4 vl = pl[q];
    float4 vr = pr[q];
    #pragma unroll
    for (int j = 0; j < 4; ++j) {
      int k = q * 4 + j;
      float fl = (&vl.x)[j];
      float fr = (&vr.x)[j];
      float w = av.x * sWe[k * 4 + 0] + av.y * sWe[k * 4 + 1] +
                av.z * sWe[k * 4 + 2] + av.w * sWe[k * 4 + 3];
      float m = fl + fr + w;
      m = (m > 0.0f) ? m : SLOPE * m;          // LeakyReLU
      logit += sAtt[k] * m;
    }
  }
  elog[e] = logit;
  atomicMaxF(&emax[d], logit);
}

// -------------------- edge pass B: exp + segment sum -----------------------
__global__ void gat_edge_den(const int* __restrict__ dst,
                             float* __restrict__ elog,
                             const float* __restrict__ emax,
                             float* __restrict__ den, int E) {
  int e = blockIdx.x * blockDim.x + threadIdx.x;
  if (e >= E) return;
  int d = dst[e];
  float ee = __expf(elog[e] - emax[d]);
  elog[e] = ee;                                // cache exp for pass C
  atomicAdd(&den[d], ee);
}

// -------------------- edge pass C: weighted scatter-add --------------------
__global__ void gat_edge_agg(const int* __restrict__ src,
                             const int* __restrict__ dst,
                             const float* __restrict__ elog,
                             const float* __restrict__ den,
                             const float* __restrict__ xl,
                             float* __restrict__ out, int E) {
  int e = blockIdx.x * blockDim.x + threadIdx.x;
  if (e >= E) return;
  int s = src[e], d = dst[e];
  const float4* pl = (const float4*)(xl + (size_t)s * 48);
  float* po = out + (size_t)d * 48;
  __builtin_prefetch(pl, 0, 0);
  float alpha = elog[e] / fmaxf(den[d], 1e-16f);
  #pragma unroll
  for (int q = 0; q < 12; ++q) {
    float4 v = pl[q];
    atomicAdd(&po[q * 4 + 0], alpha * v.x);
    atomicAdd(&po[q * 4 + 1], alpha * v.y);
    atomicAdd(&po[q * 4 + 2], alpha * v.z);
    atomicAdd(&po[q * 4 + 3], alpha * v.w);
  }
}

// -------------------- bias + ReLU ------------------------------------------
__global__ void gat_bias_relu(float* __restrict__ h,
                              const float* __restrict__ b, int total) {
  int i = blockIdx.x * blockDim.x + threadIdx.x;
  if (i >= total) return;
  float v = h[i] + b[i % 48];
  h[i] = (v > 0.0f) ? v : 0.0f;
}

// -------------------- readout: g = sum_nodes h -----------------------------
__global__ void gat_zero_vec(float* __restrict__ p, int n) {
  int i = blockIdx.x * blockDim.x + threadIdx.x;
  if (i < n) p[i] = 0.0f;
}

__global__ void gat_readout_sum(const float* __restrict__ h,
                                float* __restrict__ g, int n) {
  __shared__ float sg[48];
  if (threadIdx.x < 48) sg[threadIdx.x] = 0.0f;
  __syncthreads();
  size_t total = (size_t)n * 48;
  size_t stride = (size_t)gridDim.x * blockDim.x;
  for (size_t i = blockIdx.x * (size_t)blockDim.x + threadIdx.x;
       i < total; i += stride) {
    atomicAdd(&sg[(int)(i % 48)], h[i]);       // ds_add_f32
  }
  __syncthreads();
  if (threadIdx.x < 48) atomicAdd(&g[threadIdx.x], sg[threadIdx.x]);
}

// -------------------- FC head: fc1(relu) -> fc2 -> softmax -----------------
__global__ void gat_fc_head(const float* __restrict__ g,
                            const float* __restrict__ w1,  // [192 x 48]
                            const float* __restrict__ b1,
                            const float* __restrict__ w2,  // [10 x 192]
                            const float* __restrict__ b2,
                            float* __restrict__ out) {
  __shared__ float sg[48];
  __shared__ float sh[192];
  __shared__ float sl[10];
  if (threadIdx.x < 48) sg[threadIdx.x] = g[threadIdx.x];
  __syncthreads();
  if (threadIdx.x < 192) {
    float acc = b1[threadIdx.x];
    const float* wr = w1 + (size_t)threadIdx.x * 48;
    #pragma unroll 8
    for (int k = 0; k < 48; ++k) acc += sg[k] * wr[k];
    sh[threadIdx.x] = (acc > 0.0f) ? acc : 0.0f;
  }
  __syncthreads();
  if (threadIdx.x < 10) {
    float acc = b2[threadIdx.x];
    const float* wr = w2 + (size_t)threadIdx.x * 192;
    for (int k = 0; k < 192; ++k) acc += sh[k] * wr[k];
    sl[threadIdx.x] = acc;
  }
  __syncthreads();
  if (threadIdx.x == 0) {
    float mx = sl[0];
    for (int i = 1; i < 10; ++i) mx = fmaxf(mx, sl[i]);
    float s = 0.0f;
    for (int i = 0; i < 10; ++i) s += __expf(sl[i] - mx);
    float inv = 1.0f / s;
    for (int i = 0; i < 10; ++i) out[i] = __expf(sl[i] - mx) * inv;
  }
}

// ---------------------------------------------------------------------------
extern "C" void kernel_launch(void* const* d_in, const int* in_sizes, int n_in,
                              void* d_out, int out_size, void* d_ws, size_t ws_size,
                              hipStream_t stream) {
  (void)n_in; (void)out_size; (void)ws_size;
  const float* x  = (const float*)d_in[0];
  const int*   ei = (const int*)d_in[1];
  const float* ea = (const float*)d_in[2];
  const int N = in_sizes[0] / 128;
  const int E = in_sizes[1] / 2;
  const int* src = ei;        // edge_index[0]
  const int* dst = ei + E;    // edge_index[1]

  // workspace carve-up (floats)
  float* ws   = (float*)d_ws;
  size_t n48  = (size_t)N * 48;
  float* xl   = ws;
  float* xr   = xl + n48;
  float* h0   = xr + n48;
  float* h1   = h0 + n48;
  float* elog = h1 + n48;
  float* emax = elog + (size_t)E;
  float* den  = emax + (size_t)N;
  float* gbuf = den + (size_t)N;

  const float* hin = x;
  float* hbufs[2] = {h0, h1};

  const int tE  = (E + 255) / 256;
  const int tNF = (N * 48 + 255) / 256;
  const int mtiles = N / 16;                   // N % 16 == 0 (N = 100000)
  const int waves  = mtiles * 3;               // 48 cols = 3 N-tiles
  const int gblocks = (waves + 3) / 4;         // 4 waves / 128-thread block

  for (int l = 0; l < 3; ++l) {
    const float* Wl   = (const float*)d_in[3 + l * 7 + 0];
    const float* bl   = (const float*)d_in[3 + l * 7 + 1];
    const float* Wr   = (const float*)d_in[3 + l * 7 + 2];
    const float* br   = (const float*)d_in[3 + l * 7 + 3];
    const float* We   = (const float*)d_in[3 + l * 7 + 4];
    const float* attv = (const float*)d_in[3 + l * 7 + 5];
    const float* bout = (const float*)d_in[3 + l * 7 + 6];
    float* hout = hbufs[l & 1];

    if (l == 0) {
      gat_xform_wmma<128><<<gblocks, 128, 0, stream>>>(hin, Wl, bl, xl, mtiles);
      gat_xform_wmma<128><<<gblocks, 128, 0, stream>>>(hin, Wr, br, xr, mtiles);
    } else {
      gat_xform_wmma<48><<<gblocks, 128, 0, stream>>>(hin, Wl, bl, xl, mtiles);
      gat_xform_wmma<48><<<gblocks, 128, 0, stream>>>(hin, Wr, br, xr, mtiles);
    }

    gat_layer_init<<<tNF, 256, 0, stream>>>(hout, emax, den, N);
    gat_edge_logits<<<tE, 256, 0, stream>>>(src, dst, ea, xl, xr, We, attv,
                                            elog, emax, E);
    gat_edge_den<<<tE, 256, 0, stream>>>(dst, elog, emax, den, E);
    gat_edge_agg<<<tE, 256, 0, stream>>>(src, dst, elog, den, xl, hout, E);
    gat_bias_relu<<<tNF, 256, 0, stream>>>(hout, bout, N * 48);

    hin = hout;
  }

  const float* fc1w = (const float*)d_in[24];
  const float* fc1b = (const float*)d_in[25];
  const float* fc2w = (const float*)d_in[26];
  const float* fc2b = (const float*)d_in[27];

  gat_zero_vec<<<1, 64, 0, stream>>>(gbuf, 48);
  gat_readout_sum<<<512, 256, 0, stream>>>(hin, gbuf, N);
  gat_fc_head<<<1, 192, 0, stream>>>(gbuf, fc1w, fc1b, fc2w, fc2b,
                                     (float*)d_out);
}
